// LDM_7352984011089
// MI455X (gfx1250) — matrix operand order repaired and checked by reading, not compiled, not run on hardware
//
#include <hip/hip_runtime.h>

// Problem dimensions (from reference)
#define Y_DIM 512
#define X_DIM 256
#define R_DIM 1024
#define B_DIM 64
#define T_DIM 256

#define NB_RNN 16   // workgroups in persistent RNN kernel
#define NB_COMB 16  // workgroups in persistent comb kernel

typedef __attribute__((ext_vector_type(8)))  _Float16 v8h;
typedef __attribute__((ext_vector_type(16))) _Float16 v16h;
typedef __attribute__((ext_vector_type(8)))  float    v8f;

// ---------------------------------------------------------------------------
// WMMA fragment helpers (CDNA5 v_wmma_f32_16x16x32_f16, wave32; A/B k order is this code's own, same for both)
//
// A (16x32 f16): lane L holds row M = L&15; khi = L>>4 selects K-chunk.
//   elements e = K[16*khi + e]  (32B contiguous)
// B (32x16 f16): lane L holds column N = L&15 (== a row of W since we compute
//   x @ W.T); elements e = K[16*khi + e]  (32B contiguous)
// C/D (16x16 f32): lane L, vgpr r -> (M = r + 8*(L>>4), N = L&15)
// ---------------------------------------------------------------------------
__device__ __forceinline__ v16h load_a_frag(const _Float16* __restrict__ p, int khi) {
  return *(const v16h*)(p + 16 * khi);
}
__device__ __forceinline__ v16h load_b_frag(const _Float16* __restrict__ p, int khi) {
  return *(const v16h*)(p + 16 * khi);
}
__device__ __forceinline__ v8f wmma16(v16h a, v16h b, v8f c) {
  return __builtin_amdgcn_wmma_f32_16x16x32_f16(
      /*neg_a=*/false, a, /*neg_b=*/false, b,
      /*c_mod=*/(short)0, c, /*reuse_a=*/false, /*reuse_b=*/false);
}

// ---------------------------------------------------------------------------
// Device-scope grid barrier for persistent kernels. All NB workgroups are
// co-resident (tiny grid on a many-WGP part), so spinning is deadlock-free.
// Release/acquire at agent scope orders the L2-visible tensor writes.
// ---------------------------------------------------------------------------
__device__ __forceinline__ void grid_sync(unsigned int* cnt, unsigned int target) {
  __threadfence();      // release: flush this WG's global writes to device scope
  __syncthreads();      // all waves of the WG have arrived (and stored)
  if (threadIdx.x == 0) {
    __hip_atomic_fetch_add(cnt, 1u, __ATOMIC_ACQ_REL, __HIP_MEMORY_SCOPE_AGENT);
    while (__hip_atomic_load(cnt, __ATOMIC_ACQUIRE, __HIP_MEMORY_SCOPE_AGENT) < target)
      __builtin_amdgcn_s_sleep(1);
  }
  __syncthreads();
  __threadfence();      // acquire: invalidate stale lines before next phase reads
}

// ---------------------------------------------------------------------------
// Prep: f32 -> f16 conversion and state/counter init
// ---------------------------------------------------------------------------
__global__ void cvt_f16_kernel(const float* __restrict__ src,
                               _Float16* __restrict__ dst, int n) {
  int i = blockIdx.x * blockDim.x + threadIdx.x;
  if (i < n) dst[i] = (_Float16)src[i];
}

__global__ void init_state_kernel(const float* __restrict__ h0,
                                  const float* __restrict__ xq0,
                                  _Float16* __restrict__ h16,
                                  _Float16* __restrict__ x16,
                                  unsigned int* __restrict__ sync_rnn,
                                  unsigned int* __restrict__ sync_comb) {
  int i = blockIdx.x * blockDim.x + threadIdx.x;
  if (i < B_DIM * R_DIM) h16[i] = (_Float16)h0[i & (R_DIM - 1)];
  if (i < B_DIM * X_DIM) x16[i] = (_Float16)xq0[i & (X_DIM - 1)];
  if (i == 0) { *sync_rnn = 0u; *sync_comb = 0u; }  // re-zeroed every call
}

// ---------------------------------------------------------------------------
// Persistent RNN scan (reversed time), all 256 steps in one kernel.
// Step s: h_new = relu(h@W_hh.T + data[:,T-1-s,:]@W_ih.T + b_ih + b_hh)
// Fused K = 1024+512. Grid: 16 WG x 512 thr (16 waves); each wave owns one
// 16x16 tile: mtile = wave&3, ntile = blockIdx*4 + (wave>>2). h16 ping-pongs
// in-kernel; rnn_out[T-1-s] (f32, time-major TxBxR) is the scan output.
// ---------------------------------------------------------------------------
__global__ __launch_bounds__(512) void rnn_scan_kernel(
    _Float16* __restrict__ h16a, _Float16* __restrict__ h16b,
    const _Float16* __restrict__ Whh16, const _Float16* __restrict__ Wih16,
    const _Float16* __restrict__ data16, const float* __restrict__ b_ih,
    const float* __restrict__ b_hh, float* __restrict__ rnn_out,
    unsigned int* __restrict__ sync) {
  const int lane = threadIdx.x & 31;
  const int wave = threadIdx.x >> 5;
  const int khi  = lane >> 4;
  const int l16  = lane & 15;
  const int m0   = (wave & 3) * 16;
  const int n0   = blockIdx.x * 64 + (wave >> 2) * 16;
  const int b = m0 + l16;  // A row = batch index
  const int n = n0 + l16;  // B row = output feature

  const _Float16* __restrict__ b_rec = Whh16 + (size_t)n * R_DIM;
  const _Float16* __restrict__ b_inp = Wih16 + (size_t)n * Y_DIM;
  const float bias = b_hh[n] + b_ih[n];

  for (int s = 0; s < T_DIM; ++s) {
    const _Float16* __restrict__ h_in = (s & 1) ? h16b : h16a;
    _Float16* __restrict__ h_out      = (s & 1) ? h16a : h16b;
    const int t_rev = (T_DIM - 1) - s;

    const _Float16* __restrict__ a_h = h_in + b * R_DIM;
    const _Float16* __restrict__ a_d =
        data16 + (size_t)b * (T_DIM * Y_DIM) + (size_t)t_rev * Y_DIM;

    v8f acc = {};
#pragma unroll 4
    for (int kk = 0; kk < R_DIM; kk += 32) {
      acc = wmma16(load_a_frag(a_h + kk, khi), load_b_frag(b_rec + kk, khi), acc);
    }
#pragma unroll 4
    for (int kk = 0; kk < Y_DIM; kk += 32) {
      acc = wmma16(load_a_frag(a_d + kk, khi), load_b_frag(b_inp + kk, khi), acc);
    }

#pragma unroll
    for (int r = 0; r < 8; ++r) {
      const int m = m0 + r + 8 * khi;  // batch row of this D element
      float v = acc[r] + bias;
      v = v > 0.f ? v : 0.f;  // relu
      h_out[m * R_DIM + n0 + l16] = (_Float16)v;
      rnn_out[(size_t)t_rev * (B_DIM * R_DIM) + m * R_DIM + n0 + l16] = v;
    }

    grid_sync(sync, (unsigned)NB_RNN * (unsigned)(s + 1));
  }
}

// ---------------------------------------------------------------------------
// Persistent comb scan (forward time), all 256 steps in one kernel.
// Phase 1: h_comb = 0.5*(tanh(x_prev@Wc_xh.T + bc_xh) + rnn_out[t])
//          64x1024 = 256 tiles == one per wave (16 WG x 16 waves). K=256.
// Phase 2: mu = hc@Wc_mu.T + bc_mu ; lv = hc@Wc_lv.T + bc_lv ;
//          x = mu + eps*exp(0.5*lv).  64 mu/lv tile-pairs; each pair is
//          computed by 4 waves (K split 4 x 256) + LDS reduction, so all 256
//          waves stay busy. Grid barrier between phases and between steps.
// ---------------------------------------------------------------------------
__global__ __launch_bounds__(512) void comb_scan_kernel(
    _Float16* __restrict__ x16, const _Float16* __restrict__ Wcxh16,
    const float* __restrict__ bc_xh, const float* __restrict__ rnn_out,
    _Float16* __restrict__ hcomb16, const _Float16* __restrict__ Wcmu16,
    const _Float16* __restrict__ Wclv16, const float* __restrict__ bc_mu,
    const float* __restrict__ bc_lv, const float* __restrict__ eps,
    float* __restrict__ out_xs, float* __restrict__ out_mus,
    float* __restrict__ out_lvs, unsigned int* __restrict__ sync) {
  __shared__ float red[4][4][32][16];  // [ksplit][pair_local][lane][mu0..7,lv0..7]

  const int lane = threadIdx.x & 31;
  const int wave = threadIdx.x >> 5;
  const int khi  = lane >> 4;
  const int l16  = lane & 15;

  // ---- phase 1 tile (one 16x16 tile of h_comb per wave) ----
  const int p1_m0 = (wave & 3) * 16;
  const int p1_n0 = blockIdx.x * 64 + (wave >> 2) * 16;
  const int p1_b = p1_m0 + l16;
  const int p1_n = p1_n0 + l16;
  const _Float16* __restrict__ p1_brow = Wcxh16 + (size_t)p1_n * X_DIM;
  const float p1_bias = bc_xh[p1_n];

  // ---- phase 2 tile-pair (4 waves share a pair; K split 4x256) ----
  const int pair   = blockIdx.x * 4 + (wave & 3);  // 0..63
  const int ksplit = wave >> 2;                    // 0..3
  const int p2_m0 = (pair & 3) * 16;
  const int p2_n0 = (pair >> 2) * 16;
  const int p2_b = p2_m0 + l16;
  const int p2_n = p2_n0 + l16;
  const int k0 = ksplit * 256;
  const _Float16* __restrict__ bmu_row = Wcmu16 + (size_t)p2_n * R_DIM + k0;
  const _Float16* __restrict__ blv_row = Wclv16 + (size_t)p2_n * R_DIM + k0;
  const float bmu = bc_mu[p2_n];
  const float blv = bc_lv[p2_n];

  unsigned int gen = 0;
  for (int t = 0; t < T_DIM; ++t) {
    // ================= phase 1 =================
    {
      const _Float16* __restrict__ a_row = x16 + p1_b * X_DIM;
      v8f acc = {};
#pragma unroll
      for (int kk = 0; kk < X_DIM; kk += 32) {
        acc = wmma16(load_a_frag(a_row + kk, khi),
                     load_b_frag(p1_brow + kk, khi), acc);
      }
      const float* __restrict__ rt = rnn_out + (size_t)t * (B_DIM * R_DIM);
#pragma unroll
      for (int r = 0; r < 8; ++r) {
        const int m = p1_m0 + r + 8 * khi;
        float g = tanhf(acc[r] + p1_bias);
        float hc = 0.5f * (g + rt[m * R_DIM + p1_n0 + l16]);
        hcomb16[m * R_DIM + p1_n0 + l16] = (_Float16)hc;
      }
    }
    grid_sync(sync, (unsigned)NB_COMB * (++gen));

    // ================= phase 2 =================
    {
      const _Float16* __restrict__ a_row = hcomb16 + p2_b * R_DIM + k0;
      v8f acc_mu = {};
      v8f acc_lv = {};
#pragma unroll 4
      for (int kk = 0; kk < 256; kk += 32) {
        v16h af = load_a_frag(a_row + kk, khi);
        acc_mu = wmma16(af, load_b_frag(bmu_row + kk, khi), acc_mu);
        acc_lv = wmma16(af, load_b_frag(blv_row + kk, khi), acc_lv);
      }
#pragma unroll
      for (int r = 0; r < 8; ++r) {
        red[ksplit][wave & 3][lane][r]     = acc_mu[r];
        red[ksplit][wave & 3][lane][8 + r] = acc_lv[r];
      }
      __syncthreads();
      if (ksplit == 0) {
        const int pl = wave & 3;
#pragma unroll
        for (int r = 0; r < 8; ++r) {
          float mu = red[0][pl][lane][r] + red[1][pl][lane][r] +
                     red[2][pl][lane][r] + red[3][pl][lane][r] + bmu;
          float lv = red[0][pl][lane][8 + r] + red[1][pl][lane][8 + r] +
                     red[2][pl][lane][8 + r] + red[3][pl][lane][8 + r] + blv;
          const int m = p2_m0 + r + 8 * khi;        // batch
          const int col = p2_n0 + l16;              // x-feature
          const size_t idx =
              (size_t)m * (T_DIM * X_DIM) + (size_t)t * X_DIM + col;
          float x = mu + eps[idx] * __expf(0.5f * lv);
          out_xs[idx]  = x;
          out_mus[idx] = mu;
          out_lvs[idx] = lv;
          x16[m * X_DIM + col] = (_Float16)x;
        }
      }
      __syncthreads();  // LDS reuse safety for next step
    }
    grid_sync(sync, (unsigned)NB_COMB * (++gen));
  }
}

// ---------------------------------------------------------------------------
// Host launch
// ---------------------------------------------------------------------------
extern "C" void kernel_launch(void* const* d_in, const int* in_sizes, int n_in,
                              void* d_out, int out_size, void* d_ws,
                              size_t ws_size, hipStream_t stream) {
  (void)in_sizes; (void)n_in; (void)out_size; (void)ws_size;

  const float* data  = (const float*)d_in[0];
  const float* eps   = (const float*)d_in[1];
  const float* W_ih  = (const float*)d_in[2];
  const float* b_ih  = (const float*)d_in[3];
  const float* W_hh  = (const float*)d_in[4];
  const float* b_hh  = (const float*)d_in[5];
  const float* h0    = (const float*)d_in[6];
  const float* xq0   = (const float*)d_in[7];
  const float* Wc_xh = (const float*)d_in[8];
  const float* bc_xh = (const float*)d_in[9];
  const float* Wc_mu = (const float*)d_in[10];
  const float* bc_mu = (const float*)d_in[11];
  const float* Wc_lv = (const float*)d_in[12];
  const float* bc_lv = (const float*)d_in[13];

  char* ws = (char*)d_ws;
  size_t off = 0;
  auto take = [&](size_t bytes) -> char* {
    char* p = ws + off;
    off = (off + bytes + 255) & ~(size_t)255;
    return p;
  };

  _Float16* Whh16   = (_Float16*)take(sizeof(_Float16) * R_DIM * R_DIM);
  _Float16* Wih16   = (_Float16*)take(sizeof(_Float16) * R_DIM * Y_DIM);
  _Float16* Wcxh16  = (_Float16*)take(sizeof(_Float16) * R_DIM * X_DIM);
  _Float16* Wcmu16  = (_Float16*)take(sizeof(_Float16) * X_DIM * R_DIM);
  _Float16* Wclv16  = (_Float16*)take(sizeof(_Float16) * X_DIM * R_DIM);
  _Float16* data16  = (_Float16*)take(sizeof(_Float16) * (size_t)B_DIM * T_DIM * Y_DIM);
  _Float16* h16a    = (_Float16*)take(sizeof(_Float16) * B_DIM * R_DIM);
  _Float16* h16b    = (_Float16*)take(sizeof(_Float16) * B_DIM * R_DIM);
  _Float16* x16     = (_Float16*)take(sizeof(_Float16) * B_DIM * X_DIM);
  _Float16* hcomb16 = (_Float16*)take(sizeof(_Float16) * B_DIM * R_DIM);
  float*    rnn_out = (float*)take(sizeof(float) * (size_t)T_DIM * B_DIM * R_DIM);
  unsigned int* sync_rnn  = (unsigned int*)take(sizeof(unsigned int));
  unsigned int* sync_comb = (unsigned int*)take(sizeof(unsigned int));

  // --- prep: f16 weight/data copies (L2-resident from here on) ---
  auto cvt = [&](const float* s, _Float16* d, int n) {
    cvt_f16_kernel<<<(n + 255) / 256, 256, 0, stream>>>(s, d, n);
  };
  cvt(W_hh, Whh16, R_DIM * R_DIM);
  cvt(W_ih, Wih16, R_DIM * Y_DIM);
  cvt(Wc_xh, Wcxh16, R_DIM * X_DIM);
  cvt(Wc_mu, Wcmu16, X_DIM * R_DIM);
  cvt(Wc_lv, Wclv16, X_DIM * R_DIM);
  cvt(data, data16, B_DIM * T_DIM * Y_DIM);
  init_state_kernel<<<(B_DIM * R_DIM + 255) / 256, 256, 0, stream>>>(
      h0, xq0, h16a, x16, sync_rnn, sync_comb);

  // --- persistent RNN scan (one launch, in-kernel grid barriers) ---
  rnn_scan_kernel<<<NB_RNN, 512, 0, stream>>>(h16a, h16b, Whh16, Wih16, data16,
                                              b_ih, b_hh, rnn_out, sync_rnn);

  // --- persistent comb scan (one launch) ---
  float* out_xs  = (float*)d_out;
  float* out_mus = out_xs + (size_t)B_DIM * T_DIM * X_DIM;
  float* out_lvs = out_mus + (size_t)B_DIM * T_DIM * X_DIM;
  comb_scan_kernel<<<NB_COMB, 512, 0, stream>>>(
      x16, Wcxh16, bc_xh, rnn_out, hcomb16, Wcmu16, Wclv16, bc_mu, bc_lv, eps,
      out_xs, out_mus, out_lvs, sync_comb);
}
